// VersorAttention_12146167513836
// MI455X (gfx1250) — matrix operand order repaired and checked by reading, not compiled, not run on hardware
//
#include <hip/hip_runtime.h>

typedef __attribute__((ext_vector_type(16))) __bf16 v16bf;
typedef __attribute__((ext_vector_type(8)))  float  v8f;

#define EPS 1e-6f

static const int Bb = 2, Ss = 1024, Hh = 8;
static const int DC = 4096;   // D*32
static const int MR = 2048;   // B*S

// Cl(4,1) blade product sign: e_a * e_b = sign * e_{a^b}; METRIC = (+,+,+,+,-)
__device__ __forceinline__ float blade_sign(int a, int b) {
  int s = 0;
  int t = a >> 1;
  while (t) { s += __popc(t & b); t >>= 1; }
  if (a & b & 16) s++;           // e4*e4 = -1
  return (s & 1) ? -1.0f : 1.0f;
}

struct bf4 { __bf16 a, b, c, d; };

// ---------------- fp32 -> bf16 convert (4 elems / thread) ----------------
__global__ void cvt_f32_bf16(const float* __restrict__ in, __bf16* __restrict__ out, int n4) {
  int i = blockIdx.x * 256 + threadIdx.x;
  if (i >= n4) return;
  float4 v = ((const float4*)in)[i];
  bf4 o;
  o.a = (__bf16)v.x; o.b = (__bf16)v.y; o.c = (__bf16)v.z; o.d = (__bf16)v.w;
  ((bf4*)out)[i] = o;
}

// ------------- expand W (O,I,32) fp32 -> W_mat^T (N=O*32, K=I*32) bf16 -------------
// W_matT[o*32+k][i*32+l] = W[o,i,k^l] * sign(k^l, l)
__global__ void expand_wmatT(const float* __restrict__ W, __bf16* __restrict__ WmT) {
  size_t idx = (size_t)blockIdx.x * 256 + threadIdx.x;   // 4096*4096 total
  int rowN = (int)(idx >> 12);
  int colK = (int)(idx & 4095);
  int o = rowN >> 5, kb = rowN & 31;
  int i = colK >> 5, l = colK & 31;
  int j = kb ^ l;
  float w = W[((size_t)o * 128 + i) * 32 + j] * blade_sign(j, l);
  WmT[idx] = (__bf16)w;
}

// ---------------- unified WMMA GEMM: C(MxN) = A(MxK) * Bt(NxK)^T ----------------
// A, Bt bf16 row-major (Bt is N-major), C fp32. 256 threads, 8 waves (2M x 4N),
// block tile 128x128x32, double-buffered LDS; each wave computes 64x32 = 8 wmma tiles.
// mode 0: plain store.  mode 1: fused versor attention score transform.
union U16 { uint4 u[2]; v16bf v; };

__global__ __launch_bounds__(256) void gemm_nt_wmma(
    const __bf16* __restrict__ A, const __bf16* __restrict__ Bt, float* __restrict__ C,
    int M, int N, int K, int lda, int ldb, int ldc,
    long long aB, long long aH, long long bB, long long bH, long long cB, long long cH, int nH,
    int mode, const float* __restrict__ qn, const float* __restrict__ kn, int hnStride,
    const float* __restrict__ lamPtr, float scale) {
  __shared__ __bf16 As[2][128][40];
  __shared__ __bf16 Bs[2][128][40];

  int z  = blockIdx.z;
  int zb = z / nH, zh = z % nH;
  A  += zb * aB + zh * aH;
  Bt += zb * bB + zh * bH;
  C  += zb * cB + zh * cH;
  const float* qnp = qn ? qn + (size_t)z * hnStride : nullptr;
  const float* knp = kn ? kn + (size_t)z * hnStride : nullptr;
  float lamv = (mode == 1) ? lamPtr[0] : 0.0f;

  int tid  = threadIdx.x;
  int lane = tid & 31;          // wave32
  int wid  = tid >> 5;
  int waveM = wid >> 2;         // 0..1
  int waveN = wid & 3;          // 0..3
  int mrow = lane & 15;
  int g    = lane >> 4;

  int blockM = blockIdx.y * 128;
  int blockN = blockIdx.x * 128;

  int ldRow = tid >> 1;         // 0..127
  int ldSeg = (tid & 1) * 16;   // 0 or 16

  const __bf16* gA = A  + (size_t)(blockM + ldRow) * lda + ldSeg;
  const __bf16* gB = Bt + (size_t)(blockN + ldRow) * ldb + ldSeg;

  v8f acc[4][2];
#pragma unroll
  for (int mi = 0; mi < 4; ++mi)
#pragma unroll
    for (int ni = 0; ni < 2; ++ni)
      acc[mi][ni] = (v8f){0.f, 0.f, 0.f, 0.f, 0.f, 0.f, 0.f, 0.f};

  // ---- prologue: fill buffer 0 ----
  {
    uint4 a0 = *(const uint4*)(gA);
    uint4 a1 = *(const uint4*)(gA + 8);
    uint4 b0 = *(const uint4*)(gB);
    uint4 b1 = *(const uint4*)(gB + 8);
    *(uint4*)&As[0][ldRow][ldSeg]     = a0;
    *(uint4*)&As[0][ldRow][ldSeg + 8] = a1;
    *(uint4*)&Bs[0][ldRow][ldSeg]     = b0;
    *(uint4*)&Bs[0][ldRow][ldSeg + 8] = b1;
  }
  __syncthreads();

  int buf = 0;
  for (int k0 = 0; k0 < K; k0 += 32) {
    bool more = (k0 + 32 < K);
    uint4 a0n, a1n, b0n, b1n;
    if (more) {
      // issue next tile's global loads early; latency overlaps the wmma burst
      a0n = *(const uint4*)(gA + k0 + 32);
      a1n = *(const uint4*)(gA + k0 + 40);
      b0n = *(const uint4*)(gB + k0 + 32);
      b1n = *(const uint4*)(gB + k0 + 40);
      if (k0 + 64 < K) {
        __builtin_prefetch(gA + k0 + 64, 0, 1);   // global_prefetch_b8
        __builtin_prefetch(gB + k0 + 64, 0, 1);
      }
    }

    U16 af[4], bfr[2];
    // A fragment (16x32 bf16): lanes 0-15 rows, elems 0-7 -> K=g*8+0..7, 8-15 -> K=16+g*8+0..7
#pragma unroll
    for (int mi = 0; mi < 4; ++mi) {
      const __bf16* p = &As[buf][waveM * 64 + mi * 16 + mrow][0];
      af[mi].u[0] = *(const uint4*)(p + g * 8);
      af[mi].u[1] = *(const uint4*)(p + 16 + g * 8);
    }
    // B fragment (32x16 bf16): lane holds column n=lane&15, elems 0-7 -> K=g*8+0..7, 8-15 -> K=16+g*8+0..7 (N-major LDS)
#pragma unroll
    for (int ni = 0; ni < 2; ++ni) {
      const __bf16* p = &Bs[buf][waveN * 32 + ni * 16 + mrow][0];
      bfr[ni].u[0] = *(const uint4*)(p + g * 8);
      bfr[ni].u[1] = *(const uint4*)(p + 16 + g * 8);
    }
#pragma unroll
    for (int mi = 0; mi < 4; ++mi)
#pragma unroll
      for (int ni = 0; ni < 2; ++ni)
        acc[mi][ni] = __builtin_amdgcn_wmma_f32_16x16x32_bf16(
            false, af[mi].v, false, bfr[ni].v, (short)0, acc[mi][ni], false, false);

    if (more) {
      int nb = buf ^ 1;   // safe: all waves finished reading nb at the previous barrier
      *(uint4*)&As[nb][ldRow][ldSeg]     = a0n;
      *(uint4*)&As[nb][ldRow][ldSeg + 8] = a1n;
      *(uint4*)&Bs[nb][ldRow][ldSeg]     = b0n;
      *(uint4*)&Bs[nb][ldRow][ldSeg + 8] = b1n;
      __syncthreads();
      buf = nb;
    }
  }

  // Epilogue. C 16x16 f32 layout: lane&15 = col, row = g*8 + e.
#pragma unroll
  for (int mi = 0; mi < 4; ++mi) {
    int row0 = blockM + waveM * 64 + mi * 16 + g * 8;
#pragma unroll
    for (int ni = 0; ni < 2; ++ni) {
      int col = blockN + waveN * 32 + ni * 16 + mrow;
      float kk = (mode == 1) ? knp[col] : 0.0f;
#pragma unroll
      for (int e = 0; e < 8; ++e) {
        float sv = acc[mi][ni][e];
        int r = row0 + e;
        if (mode == 1) {
          float qq = qnp[r];
          float t = qq * kk - sv * sv;
          t = t > 0.f ? t : 0.f;
          sv = scale * sv + lamv * (scale * sqrtf(t + EPS));
        }
        C[(size_t)r * ldc + col] = sv;
      }
    }
  }
}

// --------- per-(b,s) row post-process: RMS normalize each 32-blade group ---------
// mode 0: -> bf16 q (fold SIG) + per-head |q|^2 ; mode 1: -> bf16 k + per-head |k|^2
// mode 2: -> bf16 v transposed per head (B,H,512,S) ; mode 3: -> fp32 (final output)
__global__ void postproc_rows(const float* __restrict__ in, void* __restrict__ out,
                              float* __restrict__ hn, int mode) {
  int tid = threadIdx.x;
  size_t rowBase = (size_t)blockIdx.x * DC;
  const float* rp = in + rowBase + tid * 16;
  float v[16];
#pragma unroll
  for (int e = 0; e < 4; ++e) {
    float4 f = ((const float4*)rp)[e];
    v[e * 4 + 0] = f.x; v[e * 4 + 1] = f.y; v[e * 4 + 2] = f.z; v[e * 4 + 3] = f.w;
  }
  float ss = 0.f;
#pragma unroll
  for (int e = 0; e < 16; ++e) ss += v[e] * v[e];

  __shared__ float part[256];
  __shared__ float ns[128];
  part[tid] = ss;
  __syncthreads();
  int gi = tid >> 1;
  float gs = part[gi * 2] + part[gi * 2 + 1];
  float inv = rsqrtf(gs * (1.0f / 32.0f) + EPS);
  if ((tid & 1) == 0) ns[gi] = gs * inv * inv;   // |normalized group|^2
  __syncthreads();

  int bI = blockIdx.x >> 10, sI = blockIdx.x & 1023;

  if (mode == 0) {
    __bf16* o = (__bf16*)out + rowBase + tid * 16;
#pragma unroll
    for (int e = 0; e < 16; ++e) {
      int blade = (tid & 1) * 16 + e;
      o[e] = (__bf16)(v[e] * inv * blade_sign(blade, blade));   // SIG fold
    }
  } else if (mode == 1) {
    __bf16* o = (__bf16*)out + rowBase + tid * 16;
#pragma unroll
    for (int e = 0; e < 16; ++e) o[e] = (__bf16)(v[e] * inv);
  } else if (mode == 2) {
    __bf16* o = (__bf16*)out;
    int c0 = tid * 16;
    int hh = c0 >> 9, d0 = c0 & 511;
    size_t base = ((size_t)(bI * Hh + hh) * 512 + d0) * Ss + sI;
#pragma unroll
    for (int e = 0; e < 16; ++e) o[base + (size_t)e * Ss] = (__bf16)(v[e] * inv);
  } else {
    float* o = (float*)out + rowBase + tid * 16;
#pragma unroll
    for (int e = 0; e < 16; ++e) o[e] = v[e] * inv;
  }

  if (mode <= 1 && tid < 8) {   // per-head squared norm (16 groups / head)
    float s2 = 0.f;
#pragma unroll
    for (int i = 0; i < 16; ++i) s2 += ns[tid * 16 + i];
    hn[((size_t)bI * Hh + tid) * Ss + sI] = s2;
  }
}

// ---------------- row softmax (1024 wide) -> bf16 probs ----------------
__global__ void softmax_rows(const float* __restrict__ sc, __bf16* __restrict__ pr) {
  int tid = threadIdx.x;
  size_t base = (size_t)blockIdx.x * Ss;
  float4 xv = ((const float4*)(sc + base))[tid];
  __shared__ float red[256];
  float m = fmaxf(fmaxf(xv.x, xv.y), fmaxf(xv.z, xv.w));
  red[tid] = m;
  __syncthreads();
  for (int s2 = 128; s2 > 0; s2 >>= 1) { if (tid < s2) red[tid] = fmaxf(red[tid], red[tid + s2]); __syncthreads(); }
  m = red[0];
  __syncthreads();
  float e0 = __expf(xv.x - m), e1 = __expf(xv.y - m), e2 = __expf(xv.z - m), e3 = __expf(xv.w - m);
  red[tid] = e0 + e1 + e2 + e3;
  __syncthreads();
  for (int s2 = 128; s2 > 0; s2 >>= 1) { if (tid < s2) red[tid] += red[tid + s2]; __syncthreads(); }
  float inv = 1.0f / red[0];
  __bf16* o = pr + base + tid * 4;
  o[0] = (__bf16)(e0 * inv); o[1] = (__bf16)(e1 * inv);
  o[2] = (__bf16)(e2 * inv); o[3] = (__bf16)(e3 * inv);
}

extern "C" void kernel_launch(void* const* d_in, const int* in_sizes, int n_in,
                              void* d_out, int out_size, void* d_ws, size_t ws_size,
                              hipStream_t stream) {
  (void)in_sizes; (void)n_in; (void)out_size; (void)ws_size;
  const float* x   = (const float*)d_in[0];
  const float* Wq  = (const float*)d_in[1];
  const float* Wk  = (const float*)d_in[2];
  const float* Wv  = (const float*)d_in[3];
  const float* Wo  = (const float*)d_in[4];
  const float* lam = (const float*)d_in[5];
  float* out = (float*)d_out;
  char* ws = (char*)d_ws;

  size_t off = 0;
  __bf16* xb   = (__bf16*)(ws + off); off += (size_t)MR * DC * 2;       // also reused as attn bf16
  __bf16* wmt  = (__bf16*)(ws + off); off += (size_t)DC * DC * 2;
  float*  tmp  = (float*)(ws + off);  off += (size_t)MR * DC * 4;
  __bf16* qb   = (__bf16*)(ws + off); off += (size_t)MR * DC * 2;
  __bf16* kb   = (__bf16*)(ws + off); off += (size_t)MR * DC * 2;
  __bf16* vbT  = (__bf16*)(ws + off); off += (size_t)MR * DC * 2;
  float*  qn   = (float*)(ws + off);  off += (size_t)Bb * Hh * Ss * 4;
  float*  kn   = (float*)(ws + off);  off += (size_t)Bb * Hh * Ss * 4;
  float*  scores = (float*)(ws + off); off += (size_t)Bb * Hh * Ss * Ss * 4;
  __bf16* probs  = (__bf16*)(ws + off); off += (size_t)Bb * Hh * Ss * Ss * 2;

  dim3 thr(256);
  const int n4 = MR * DC / 4;
  const int cvtBlocks = n4 / 256;
  const int expBlocks = (DC * DC) / 256;

  // x -> bf16
  cvt_f32_bf16<<<cvtBlocks, thr, 0, stream>>>(x, xb, n4);

  // Q / K / V projections: expand W_mat^T, GEMM, normalize(+extras)
  const float* Ws[3] = {Wq, Wk, Wv};
  void* outs[3] = {qb, kb, vbT};
  float* hns[3] = {qn, kn, nullptr};
  for (int p = 0; p < 3; ++p) {
    expand_wmatT<<<expBlocks, thr, 0, stream>>>(Ws[p], wmt);
    gemm_nt_wmma<<<dim3(DC / 128, MR / 128, 1), thr, 0, stream>>>(
        xb, wmt, tmp, MR, DC, DC, DC, DC, DC,
        0, 0, 0, 0, 0, 0, 1, 0, nullptr, nullptr, 0, nullptr, 1.f);
    postproc_rows<<<MR, thr, 0, stream>>>(tmp, outs[p], hns[p], p);
  }

  // scores = scale * q.k + lam*scale*torque   (batched over B*H, fused epilogue)
  gemm_nt_wmma<<<dim3(Ss / 128, Ss / 128, Bb * Hh), thr, 0, stream>>>(
      qb, kb, scores, Ss, Ss, 512, DC, DC, Ss,
      (long long)Ss * DC, 512, (long long)Ss * DC, 512,
      (long long)Hh * Ss * Ss, (long long)Ss * Ss, Hh,
      1, qn, kn, Ss, lam, 0.25f);

  softmax_rows<<<Bb * Hh * Ss, thr, 0, stream>>>(scores, probs);

  // out = probs @ v   (v pre-transposed per head -> N-major B)
  gemm_nt_wmma<<<dim3(512 / 128, Ss / 128, Bb * Hh), thr, 0, stream>>>(
      probs, vbT, tmp, Ss, 512, Ss, Ss, Ss, DC,
      (long long)Hh * Ss * Ss, (long long)Ss * Ss,
      (long long)Hh * 512 * Ss, (long long)512 * Ss,
      (long long)Ss * DC, 512, Hh,
      0, nullptr, nullptr, 0, nullptr, 1.f);

  // attn out -> bf16 (reuse xb), O projection, final normalize -> d_out
  cvt_f32_bf16<<<cvtBlocks, thr, 0, stream>>>(tmp, xb, n4);
  expand_wmatT<<<expBlocks, thr, 0, stream>>>(Wo, wmt);
  gemm_nt_wmma<<<dim3(DC / 128, MR / 128, 1), thr, 0, stream>>>(
      xb, wmt, tmp, MR, DC, DC, DC, DC, DC,
      0, 0, 0, 0, 0, 0, 1, 0, nullptr, nullptr, 0, nullptr, 1.f);
  postproc_rows<<<MR, thr, 0, stream>>>(tmp, out, nullptr, 3);
}
